// VRNN_48026324303928
// MI455X (gfx1250) — matrix operand order repaired and checked by reading, not compiled, not run on hardware
//
#include <hip/hip_runtime.h>
#include <hip/hip_bf16.h>

// ---------------------------------------------------------------------------
// VRNN forward, persistent-kernel design for MI455X (gfx1250, wave32, WMMA).
//   grid = 16 blocks, each owns a 16-row batch tile and runs all T=512 steps.
//   All activations live in LDS (~197 KB of the 320 KB/WGP budget).
//   Weights packed once to bf16 in WMMA-fragment order in d_ws (L2-resident);
//   hot loop B fragments are single 32B/lane coalesced loads, A fragments are
//   ds_load_b128 pairs, GEMMs are v_wmma_f32_16x16x32_bf16.
// ---------------------------------------------------------------------------

typedef __attribute__((ext_vector_type(16))) __bf16 v16bf;
typedef __attribute__((ext_vector_type(8)))  __bf16 v8bf;
typedef __attribute__((ext_vector_type(8)))  float  v8f;
typedef __attribute__((ext_vector_type(4)))  float  v4f;
typedef __attribute__((ext_vector_type(2)))  float  v2f;

#define BATCH 256
#define TLEN  512
#define XD    128
#define XFT   256
#define HD    512
#define ZD    128
#define ZFT   256
#define NOUT  (BATCH*TLEN*ZD)
#define BDIM  256
#define LOG2PI_F 1.8378770664093453f

#define SMEM_BYTES 196640

// packed-weight indices into VParams::pk
enum { PW_PHIX=0, PW_PR, PW_PRM, PW_PRS, PW_ENC, PW_ENCM, PW_ENCS,
       PW_PHIZ, PW_DEC, PW_DECM, PW_DECS, PW_IH, PW_HH, PW_COUNT };

struct VParams {
  const float *x, *eps;
  const float *W_phix, *b_phix, *W_pr, *b_pr, *W_prm, *b_prm, *W_prs, *b_prs;
  const float *W_enc, *b_enc, *W_encm, *b_encm, *W_encs, *b_encs;
  const float *W_phiz, *b_phiz, *W_dec, *b_dec, *W_decm, *b_decm, *W_decs, *b_decs;
  const float *W_ih, *b_ih, *W_hh, *b_hh;
  const __bf16* pk[PW_COUNT];
  float *out;  // [B,T,ZD] latents, then kld scalar, then nll scalar
};

__device__ __forceinline__ float softplus_f(float x) {
  return (x > 20.f) ? x : __logf(1.f + __expf(x));
}
__device__ __forceinline__ float sigmoid_f(float x) {
  return 1.f / (1.f + __expf(-x));
}
__device__ __forceinline__ float tanh_f(float x) {
  x = fminf(fmaxf(x, -15.f), 15.f);
  const float e2 = __expf(2.f * x);
  return (e2 - 1.f) / (e2 + 1.f);
}
__device__ __forceinline__ unsigned pack_bf16x2(float a, float b) {
  union { __bf16 h[2]; unsigned u; } cv;
  cv.h[0] = (__bf16)a; cv.h[1] = (__bf16)b;
  return cv.u;
}

// A fragment (16x32 bf16) from LDS, row-major [16][ldA].
// lanes 0-15 row M=lane, K 0-15 (v0-7);
// lanes 16-31 same rows, K 16-31.
__device__ __forceinline__ v16bf load_a_lds(const __bf16* __restrict__ A,
                                            int ldA, int lane, int k0) {
  const int row  = lane & 15;
  const int half = lane >> 4;
  const __bf16* ap = A + row * ldA + k0 + (half << 4);
  v8bf lo = *(const v8bf*)(ap);        // ds_load_b128
  v8bf hi = *(const v8bf*)(ap + 8);    // ds_load_b128
  return __builtin_shufflevector(lo, hi, 0, 1, 2, 3, 4, 5, 6, 7,
                                 8, 9, 10, 11, 12, 13, 14, 15);
}

// Fallback B fragment from fp32 weight row (out column) in global.
__device__ __forceinline__ v16bf load_b_f32(const float* __restrict__ Wrow,
                                            int half) {
  const float* bp = Wrow + (half << 4);
  v16bf b;
#pragma unroll
  for (int j = 0; j < 4; ++j) {
    v4f f = *(const v4f*)(bp + 4 * j);  // global_load_b128
#pragma unroll
    for (int i = 0; i < 4; ++i) b[4 * j + i] = (__bf16)f[i];
  }
  return b;
}

// One 16xN GEMM: out[16,N] = act( A[16,K] @ W[N,K]^T + bias ).
// Waves split the N/16 tiles; K accumulated via wmma_f32_16x16x32_bf16.
// PK: B fragments come pre-packed (per tile: [K/32][32 lanes][16 bf16]).
template <int ACT, bool OUTBF, bool PK>
__device__ __forceinline__ void gemm16(const __bf16* __restrict__ A, int K,
                                       const float* __restrict__ W,
                                       const __bf16* __restrict__ Wpk,
                                       const float* __restrict__ bias, int N,
                                       void* __restrict__ outp, int wv, int lane) {
  const int half = lane >> 4;
  const int nidx = lane & 15;
  const int nkb  = K >> 5;
  for (int tIdx = wv; tIdx < (N >> 4); tIdx += 8) {
    const int col = (tIdx << 4) + nidx;
    const float bv = bias[col];
    v8f acc;
#pragma unroll
    for (int r = 0; r < 8; ++r) acc[r] = bv;
    if (PK) {
      const __bf16* Wt = Wpk + (size_t)tIdx * nkb * 512 + lane * 16;
      __builtin_prefetch(Wt, 0, 1);  // global_prefetch_b8
      for (int b = 0; b < nkb; ++b) {
        v16bf af = load_a_lds(A, K, lane, b << 5);
        v16bf bf = *(const v16bf*)(Wt + b * 512);  // 32B/lane, coalesced
        acc = __builtin_amdgcn_wmma_f32_16x16x32_bf16(false, af, false, bf,
                                                      (short)0, acc, false, false);
      }
    } else {
      const float* Wrow = W + (size_t)col * K;
      __builtin_prefetch(Wrow, 0, 1);
      for (int k = 0; k < K; k += 32) {
        v16bf af = load_a_lds(A, K, lane, k);
        v16bf bf = load_b_f32(Wrow + k, half);
        acc = __builtin_amdgcn_wmma_f32_16x16x32_bf16(false, af, false, bf,
                                                      (short)0, acc, false, false);
      }
    }
#pragma unroll
    for (int r = 0; r < 8; ++r) {
      float v = acc[r];
      if (ACT == 1) v = fmaxf(v, 0.f);
      if (ACT == 2) v = softplus_f(v);
      const int m = (half << 3) + r;
      if (OUTBF) ((__bf16*)outp)[m * N + col] = (__bf16)v;
      else       ((float*)outp)[m * N + col]  = v;
    }
  }
}

// Fused GRU: per 16-col tile compute all six gate GEMMs (gi_r/z/n, gh_r/z/n)
// in registers, apply gate nonlinearities, write h_new (bf16) to Hn.
template <bool PK>
__device__ __forceinline__ void gru_step(
    const __bf16* __restrict__ CAT2,  // [16][512] = [phi_x|phi_z]
    const __bf16* __restrict__ Hc,    // [16][512]
    __bf16* __restrict__ Hn,
    const float* __restrict__ Wih, const float* __restrict__ bih,
    const float* __restrict__ Whh, const float* __restrict__ bhh,
    const __bf16* __restrict__ pkIH, const __bf16* __restrict__ pkHH,
    int wv, int lane) {
  const int half = lane >> 4;
  const int nidx = lane & 15;
  for (int tIdx = wv; tIdx < (HD >> 4); tIdx += 8) {
    const int col = (tIdx << 4) + nidx;
    v8f aIR, aIZ, aIN, aHR, aHZ, aHN;
    const float bIR = bih[col], bIZ = bih[HD + col], bIN = bih[2 * HD + col];
    const float bHR = bhh[col], bHZ = bhh[HD + col], bHN = bhh[2 * HD + col];
#pragma unroll
    for (int r = 0; r < 8; ++r) {
      aIR[r] = bIR; aIZ[r] = bIZ; aIN[r] = bIN;
      aHR[r] = bHR; aHZ[r] = bHZ; aHN[r] = bHN;
    }
    if (PK) {
      // packed tile stride for K=512: 16 kb * 512 = 8192 elements
      const __bf16* TiR = pkIH + (size_t)tIdx * 8192 + lane * 16;
      const __bf16* TiZ = TiR + (size_t)32 * 8192;
      const __bf16* TiN = TiR + (size_t)64 * 8192;
      const __bf16* ThR = pkHH + (size_t)tIdx * 8192 + lane * 16;
      const __bf16* ThZ = ThR + (size_t)32 * 8192;
      const __bf16* ThN = ThR + (size_t)64 * 8192;
      __builtin_prefetch(TiR, 0, 1);
      __builtin_prefetch(ThR, 0, 1);
      for (int b = 0; b < 16; ++b) {
        v16bf a = load_a_lds(CAT2, HD, lane, b << 5);
        aIR = __builtin_amdgcn_wmma_f32_16x16x32_bf16(false, a, false, *(const v16bf*)(TiR + b * 512), (short)0, aIR, false, false);
        aIZ = __builtin_amdgcn_wmma_f32_16x16x32_bf16(false, a, false, *(const v16bf*)(TiZ + b * 512), (short)0, aIZ, false, false);
        aIN = __builtin_amdgcn_wmma_f32_16x16x32_bf16(false, a, false, *(const v16bf*)(TiN + b * 512), (short)0, aIN, false, false);
      }
      for (int b = 0; b < 16; ++b) {
        v16bf a = load_a_lds(Hc, HD, lane, b << 5);
        aHR = __builtin_amdgcn_wmma_f32_16x16x32_bf16(false, a, false, *(const v16bf*)(ThR + b * 512), (short)0, aHR, false, false);
        aHZ = __builtin_amdgcn_wmma_f32_16x16x32_bf16(false, a, false, *(const v16bf*)(ThZ + b * 512), (short)0, aHZ, false, false);
        aHN = __builtin_amdgcn_wmma_f32_16x16x32_bf16(false, a, false, *(const v16bf*)(ThN + b * 512), (short)0, aHN, false, false);
      }
    } else {
      const float* WiR = Wih + (size_t)col * HD;
      const float* WiZ = Wih + (size_t)(HD + col) * HD;
      const float* WiN = Wih + (size_t)(2 * HD + col) * HD;
      const float* WhR = Whh + (size_t)col * HD;
      const float* WhZ = Whh + (size_t)(HD + col) * HD;
      const float* WhN = Whh + (size_t)(2 * HD + col) * HD;
      for (int k = 0; k < HD; k += 32) {
        v16bf a = load_a_lds(CAT2, HD, lane, k);
        aIR = __builtin_amdgcn_wmma_f32_16x16x32_bf16(false, a, false, load_b_f32(WiR + k, half), (short)0, aIR, false, false);
        aIZ = __builtin_amdgcn_wmma_f32_16x16x32_bf16(false, a, false, load_b_f32(WiZ + k, half), (short)0, aIZ, false, false);
        aIN = __builtin_amdgcn_wmma_f32_16x16x32_bf16(false, a, false, load_b_f32(WiN + k, half), (short)0, aIN, false, false);
      }
      for (int k = 0; k < HD; k += 32) {
        v16bf a = load_a_lds(Hc, HD, lane, k);
        aHR = __builtin_amdgcn_wmma_f32_16x16x32_bf16(false, a, false, load_b_f32(WhR + k, half), (short)0, aHR, false, false);
        aHZ = __builtin_amdgcn_wmma_f32_16x16x32_bf16(false, a, false, load_b_f32(WhZ + k, half), (short)0, aHZ, false, false);
        aHN = __builtin_amdgcn_wmma_f32_16x16x32_bf16(false, a, false, load_b_f32(WhN + k, half), (short)0, aHN, false, false);
      }
    }
#pragma unroll
    for (int r = 0; r < 8; ++r) {
      const int m = (half << 3) + r;
      const float hv = (float)Hc[m * HD + col];
      const float rg = sigmoid_f(aIR[r] + aHR[r]);
      const float zg = sigmoid_f(aIZ[r] + aHZ[r]);
      const float ng = tanh_f(aIN[r] + rg * aHN[r]);
      Hn[m * HD + col] = (__bf16)((1.f - zg) * ng + zg * hv);
    }
  }
}

// Pack one [N,K] fp32 weight matrix into bf16 WMMA-fragment order:
// dst[((tile*(K/32)+kb)*32 + lane)*16 + i] = W[tile*16 + (lane&15)][kb*32 + 16*(lane>>4) + i]
__global__ void pack_w_kernel(const float* __restrict__ src,
                              __bf16* __restrict__ dst, int N, int K) {
  const int total = N * K;
  const int nkb = K >> 5;
  for (int p = blockIdx.x * blockDim.x + threadIdx.x; p < total;
       p += gridDim.x * blockDim.x) {
    const int i    = p & 15;
    const int lane = (p >> 4) & 31;
    const int kb   = (p >> 9) % nkb;
    const int tile = p / (nkb << 9);
    const int row  = (tile << 4) + (lane & 15);
    const int kk   = (kb << 5) + ((lane >> 4) << 4) + i;
    dst[p] = (__bf16)src[(size_t)row * K + kk];
  }
}

__global__ void init_acc_kernel(float* out) {
  if (threadIdx.x < 2) out[NOUT + threadIdx.x] = 0.f;
}

template <bool PK>
__global__ __launch_bounds__(BDIM, 1)
void vrnn_persistent_kernel(VParams p) {
  extern __shared__ __align__(16) char smem[];
  __bf16* H0  = (__bf16*)smem;             // 16x512 bf16 (h, parity 0)
  __bf16* H1  = H0 + 16 * HD;              // 16x512 bf16 (h, parity 1)
  __bf16* XB  = H1 + 16 * HD;              // 16x128 bf16 x_t
  float*  XF  = (float*)(XB + 16 * XD);    // 16x128 f32  x_t
  __bf16* PX  = (__bf16*)(XF + 16 * XD);   // 16x256 phi_x
  __bf16* PZ  = PX + 16 * XFT;             // 16x256 phi_z
  __bf16* CAT = PZ + 16 * ZFT;             // 16x768 concat scratch
  __bf16* PR  = CAT + 16 * 768;            // 16x512 prior feat
  __bf16* ENC = PR + 16 * HD;              // 16x512 enc feat
  __bf16* DEC = ENC + 16 * HD;             // 16x512 dec feat
  float*  PM  = (float*)(DEC + 16 * HD);   // 16x128 prior mean
  float*  PS  = PM + 16 * ZD;              // 16x128 prior std
  float*  EM  = PS + 16 * ZD;              // 16x128 enc mean
  float*  ES  = EM + 16 * ZD;              // 16x128 enc std
  float*  DM  = ES + 16 * ZD;              // 16x128 dec mean
  float*  DSD = DM + 16 * XD;              // 16x128 dec std
  float*  ZF  = DSD + 16 * XD;             // 16x128 z (f32)
  __bf16* ZB  = (__bf16*)(ZF + 16 * ZD);   // 16x128 z (bf16)
  float*  RED = (float*)(ZB + 16 * ZD);    // reduction scratch

  const int tid  = threadIdx.x;
  const int wv   = tid >> 5;
  const int lane = tid & 31;
  const int rb   = blockIdx.x << 4;  // batch row base of this block's tile

  // h0 = 0, 32-bit stores
  for (int i = tid; i < 16 * HD / 2; i += BDIM) ((unsigned*)H0)[i] = 0u;
  float kacc = 0.f, nacc = 0.f;
  __syncthreads();

  for (int t = 0; t < TLEN; ++t) {
    const __bf16* Hc = (t & 1) ? H1 : H0;
    __bf16*       Hn = (t & 1) ? H0 : H1;

    // ---- stage x_t tile (paired: 64-bit global, packed bf16 LDS) -------
    for (int i = tid; i < 16 * (XD / 2); i += BDIM) {
      const int m = i >> 6, k2 = i & 63;
      const v2f f = *(const v2f*)(p.x + ((size_t)(rb + m) * TLEN + t) * XD + 2 * k2);
      *(v2f*)(XF + m * XD + 2 * k2) = f;
      ((unsigned*)XB)[i] = pack_bf16x2(f.x, f.y);
    }
    __syncthreads();

    // ---- phi_x, prior feature ------------------------------------------
    gemm16<1, true, PK>(XB, XD, p.W_phix, p.pk[PW_PHIX], p.b_phix, XFT, PX, wv, lane);
    gemm16<1, true, PK>(Hc, HD, p.W_pr, p.pk[PW_PR], p.b_pr, HD, PR, wv, lane);
    __syncthreads();

    // ---- prior mean/std + build [phi_x | h] (32-bit pair copies) -------
    for (int i = tid; i < 16 * (768 / 2); i += BDIM) {
      const int m = i / 384, c = (i - m * 384) * 2;
      ((unsigned*)CAT)[i] = (c < XFT)
          ? *(const unsigned*)(PX + m * XFT + c)
          : *(const unsigned*)(Hc + m * HD + (c - XFT));
    }
    gemm16<0, false, PK>(PR, HD, p.W_prm, p.pk[PW_PRM], p.b_prm, ZD, PM, wv, lane);
    gemm16<2, false, PK>(PR, HD, p.W_prs, p.pk[PW_PRS], p.b_prs, ZD, PS, wv, lane);
    __syncthreads();

    // ---- encoder --------------------------------------------------------
    gemm16<1, true, PK>(CAT, 768, p.W_enc, p.pk[PW_ENC], p.b_enc, HD, ENC, wv, lane);
    __syncthreads();
    gemm16<0, false, PK>(ENC, HD, p.W_encm, p.pk[PW_ENCM], p.b_encm, ZD, EM, wv, lane);
    gemm16<2, false, PK>(ENC, HD, p.W_encs, p.pk[PW_ENCS], p.b_encs, ZD, ES, wv, lane);
    __syncthreads();

    // ---- rsample z, emit latent, accumulate KL (paired) ----------------
    for (int i = tid; i < 16 * (ZD / 2); i += BDIM) {
      const int m = i >> 6, k2 = (i & 63) * 2;
      const int li = m * ZD + k2;
      const v2f e  = *(const v2f*)(p.eps + ((size_t)t * BATCH + (rb + m)) * ZD + k2);
      const v2f em = *(const v2f*)(EM + li), es = *(const v2f*)(ES + li);
      const v2f pm = *(const v2f*)(PM + li), ps = *(const v2f*)(PS + li);
      v2f z; z.x = em.x + es.x * e.x; z.y = em.y + es.y * e.y;
      *(v2f*)(ZF + li) = z;
      ((unsigned*)ZB)[i] = pack_bf16x2(z.x, z.y);
      *(v2f*)(p.out + ((size_t)(rb + m) * TLEN + t) * ZD + k2) = z;
      const float d0 = em.x - pm.x, d1 = em.y - pm.y;
      kacc += __logf(ps.x / es.x) + (es.x * es.x + d0 * d0) / (2.f * ps.x * ps.x) - 0.5f;
      kacc += __logf(ps.y / es.y) + (es.y * es.y + d1 * d1) / (2.f * ps.y * ps.y) - 0.5f;
    }
    __syncthreads();

    // ---- phi_z ----------------------------------------------------------
    gemm16<1, true, PK>(ZB, ZD, p.W_phiz, p.pk[PW_PHIZ], p.b_phiz, ZFT, PZ, wv, lane);
    __syncthreads();

    // ---- decoder: build [phi_z | h] ------------------------------------
    for (int i = tid; i < 16 * (768 / 2); i += BDIM) {
      const int m = i / 384, c = (i - m * 384) * 2;
      ((unsigned*)CAT)[i] = (c < ZFT)
          ? *(const unsigned*)(PZ + m * ZFT + c)
          : *(const unsigned*)(Hc + m * HD + (c - ZFT));
    }
    __syncthreads();
    gemm16<1, true, PK>(CAT, 768, p.W_dec, p.pk[PW_DEC], p.b_dec, HD, DEC, wv, lane);
    __syncthreads();
    gemm16<0, false, PK>(DEC, HD, p.W_decm, p.pk[PW_DECM], p.b_decm, XD, DM, wv, lane);
    gemm16<2, false, PK>(DEC, HD, p.W_decs, p.pk[PW_DECS], p.b_decs, XD, DSD, wv, lane);
    __syncthreads();

    // ---- NLL + build [phi_x | phi_z] for GRU (paired) ------------------
    for (int i = tid; i < 16 * (XD / 2); i += BDIM) {
      const int li = (i >> 6) * XD + (i & 63) * 2;
      const v2f xv = *(const v2f*)(XF + li);
      const v2f dm = *(const v2f*)(DM + li);
      const v2f ds = *(const v2f*)(DSD + li);
      const float d0 = xv.x - dm.x, d1 = xv.y - dm.y;
      nacc += d0 * d0 / (2.f * ds.x * ds.x) + __logf(ds.x) + 0.5f * LOG2PI_F;
      nacc += d1 * d1 / (2.f * ds.y * ds.y) + __logf(ds.y) + 0.5f * LOG2PI_F;
    }
    for (int i = tid; i < 16 * (HD / 2); i += BDIM) {
      const int m = i >> 8, c = (i & 255) * 2;
      ((unsigned*)CAT)[i] = (c < XFT)
          ? *(const unsigned*)(PX + m * XFT + c)
          : *(const unsigned*)(PZ + m * ZFT + (c - XFT));
    }
    __syncthreads();

    // ---- fused GRU ------------------------------------------------------
    gru_step<PK>(CAT, Hc, Hn, p.W_ih, p.b_ih, p.W_hh, p.b_hh,
                 p.pk[PW_IH], p.pk[PW_HH], wv, lane);
    __syncthreads();
  }

  // ---- block-reduce KL / NLL and publish (mean over batch) -------------
  if (tid == 0) { RED[0] = 0.f; RED[1] = 0.f; }
  __syncthreads();
  atomicAdd(&RED[0], kacc);
  atomicAdd(&RED[1], nacc);
  __syncthreads();
  if (tid == 0) {
    atomicAdd(p.out + NOUT,     RED[0] * (1.f / (float)BATCH));
    atomicAdd(p.out + NOUT + 1, RED[1] * (1.f / (float)BATCH));
  }
}

extern "C" void kernel_launch(void* const* d_in, const int* in_sizes, int n_in,
                              void* d_out, int out_size, void* d_ws, size_t ws_size,
                              hipStream_t stream) {
  (void)in_sizes; (void)n_in; (void)out_size;
  VParams p;
  p.x      = (const float*)d_in[0];
  p.eps    = (const float*)d_in[1];
  p.W_phix = (const float*)d_in[2];  p.b_phix = (const float*)d_in[3];
  p.W_pr   = (const float*)d_in[4];  p.b_pr   = (const float*)d_in[5];
  p.W_prm  = (const float*)d_in[6];  p.b_prm  = (const float*)d_in[7];
  p.W_prs  = (const float*)d_in[8];  p.b_prs  = (const float*)d_in[9];
  p.W_enc  = (const float*)d_in[10]; p.b_enc  = (const float*)d_in[11];
  p.W_encm = (const float*)d_in[12]; p.b_encm = (const float*)d_in[13];
  p.W_encs = (const float*)d_in[14]; p.b_encs = (const float*)d_in[15];
  p.W_phiz = (const float*)d_in[16]; p.b_phiz = (const float*)d_in[17];
  p.W_dec  = (const float*)d_in[18]; p.b_dec  = (const float*)d_in[19];
  p.W_decm = (const float*)d_in[20]; p.b_decm = (const float*)d_in[21];
  p.W_decs = (const float*)d_in[22]; p.b_decs = (const float*)d_in[23];
  p.W_ih   = (const float*)d_in[24]; p.b_ih   = (const float*)d_in[25];
  p.W_hh   = (const float*)d_in[26]; p.b_hh   = (const float*)d_in[27];
  p.out    = (float*)d_out;

  // Weight matrices (N, K) and their d_in slots, in pk[] order.
  static const int WN[PW_COUNT]   = {256, 512, 128, 128, 512, 128, 128,
                                     256, 512, 128, 128, 1536, 1536};
  static const int WK[PW_COUNT]   = {128, 512, 512, 512, 768, 512, 512,
                                     128, 768, 512, 512, 512, 512};
  static const int WSLOT[PW_COUNT] = {2, 4, 6, 8, 10, 12, 14, 16, 18, 20, 22, 24, 26};

  size_t need = 0;
  for (int i = 0; i < PW_COUNT; ++i) need += (size_t)WN[i] * WK[i];
  const bool use_pk = (ws_size >= need * sizeof(__bf16));

  hipLaunchKernelGGL(init_acc_kernel, dim3(1), dim3(64), 0, stream, (float*)d_out);

  if (use_pk) {
    __bf16* base = (__bf16*)d_ws;
    size_t off = 0;
    for (int i = 0; i < PW_COUNT; ++i) {
      p.pk[i] = base + off;
      const int total = WN[i] * WK[i];
      const int grid = (total + BDIM - 1) / BDIM;
      hipLaunchKernelGGL(pack_w_kernel, dim3(grid), dim3(BDIM), 0, stream,
                         (const float*)d_in[WSLOT[i]], base + off, WN[i], WK[i]);
      off += (size_t)total;
    }
    hipLaunchKernelGGL((vrnn_persistent_kernel<true>), dim3(BATCH / 16),
                       dim3(BDIM), SMEM_BYTES, stream, p);
  } else {
    for (int i = 0; i < PW_COUNT; ++i) p.pk[i] = nullptr;
    hipLaunchKernelGGL((vrnn_persistent_kernel<false>), dim3(BATCH / 16),
                       dim3(BDIM), SMEM_BYTES, stream, p);
  }
}
